// MultiHeadAttention3_549755814010
// MI455X (gfx1250) — matrix operand order repaired and checked by reading, not compiled, not run on hardware
//
#include <hip/hip_runtime.h>
#include <hip/hip_bf16.h>

// ---------------------------------------------------------------------------
// Types for WMMA fragments
// ---------------------------------------------------------------------------
typedef __attribute__((ext_vector_type(16))) __bf16 v16bf;
typedef __attribute__((ext_vector_type(8)))  float  v8f;

struct Blob32 { uint4 lo, hi; };  // 32 bytes == one v16bf fragment

static __device__ __forceinline__ v16bf load_frag(const unsigned short* __restrict__ p0,
                                                  const unsigned short* __restrict__ p1) {
  Blob32 s;
  s.lo = *reinterpret_cast<const uint4*>(p0);   // fragment elements 0..7
  s.hi = *reinterpret_cast<const uint4*>(p1);   // fragment elements 8..15
  return __builtin_bit_cast(v16bf, s);
}

// float -> bf16 with round-to-nearest-even
static __device__ __forceinline__ unsigned short f2bf(float f) {
  unsigned int u = __float_as_uint(f);
  u += 0x7fffu + ((u >> 16) & 1u);
  return (unsigned short)(u >> 16);
}

#define D 640
#define NUM_B 2048
#define LEN_Q 32

// ---------------------------------------------------------------------------
// Kernel 1: per-batch  T[b,:] = bf16( sum(v[b,:]) * v[b,:] + k[b,:] )
// one wave per batch, 8 waves per block
// ---------------------------------------------------------------------------
__global__ void prep_T(const float* __restrict__ k, const float* __restrict__ v,
                       unsigned short* __restrict__ T) {
  const int lane = threadIdx.x & 31;
  const int wave = threadIdx.x >> 5;
  const int b    = blockIdx.x * 8 + wave;           // grid = 256 blocks -> 2048 batches
  const float* vb = v + b * D;
  const float* kb = k + b * D;
  float s = 0.f;
  #pragma unroll
  for (int i = 0; i < D / 32; ++i) s += vb[lane + i * 32];
  #pragma unroll
  for (int off = 16; off; off >>= 1) s += __shfl_xor(s, off, 32);
  unsigned short* tb = T + b * D;
  #pragma unroll
  for (int i = 0; i < D / 32; ++i) {
    int j = lane + i * 32;
    tb[j] = f2bf(fmaf(s, vb[j], kb[j]));
  }
}

// ---------------------------------------------------------------------------
// Kernel 2: bf16 copy of w_fc (row-major [n][k], n = output feature)
// ---------------------------------------------------------------------------
__global__ void conv_W(const float* __restrict__ w, unsigned short* __restrict__ Wb) {
  int i = blockIdx.x * blockDim.x + threadIdx.x;    // grid covers exactly 640*640
  Wb[i] = f2bf(w[i]);
}

// ---------------------------------------------------------------------------
// Kernel 3: Y(2048x640) = T(2048x640) @ Wb^T(640x640) + b_fc
// wave computes one 16x16 tile; 20 x v_wmma_f32_16x16x32_bf16 along K.
// A fragment (16-bit A 16x32, ISA 7.12.2): lane half h, elems 0..7 -> K = h*8+0..7,
//   elems 8..15 -> K = 16+h*8+0..7  (two contiguous 16B runs).
// B fragment (32x16): same K order as A: lane half h, elems 0..7 -> K = h*8+0..7,
//   elems 8..15 -> K = 16+h*8+0..7  (two contiguous 16B runs).
// C/D: lane col = lane%16; vgpr r -> row r + 8*(lane/16).
// ---------------------------------------------------------------------------
__global__ void gemm_bf16(const unsigned short* __restrict__ T,
                          const unsigned short* __restrict__ W,
                          const float* __restrict__ bias,
                          float* __restrict__ Y) {
  const int lane = threadIdx.x & 31;
  const int wave = threadIdx.x >> 5;
  const int tile = blockIdx.x * 8 + wave;           // grid = 640 blocks -> 5120 tiles
  const int tN = tile % (D / 16);
  const int tM = tile / (D / 16);
  const int half = lane >> 4;
  const int lmod = lane & 15;

  const int arow = tM * 16 + lmod;                  // batch row of T
  const int bcol = tN * 16 + lmod;                  // output feature column
  const unsigned short* aBase = T + arow * D + half * 8;
  const unsigned short* bBase = W + bcol * D + half * 8;

  v8f c = {};
  #pragma unroll
  for (int k0 = 0; k0 < D; k0 += 32) {
    v16bf a = load_frag(aBase + k0, aBase + k0 + 16);
    v16bf b = load_frag(bBase + k0, bBase + k0 + 16);
    c = __builtin_amdgcn_wmma_f32_16x16x32_bf16(false, a, false, b,
                                                (short)0, c, false, false);
  }

  const float bv = bias[bcol];
  float* yBase = Y + (tM * 16) * D + bcol;          // this lane's column
  #pragma unroll
  for (int r = 0; r < 8; ++r) {
    int m = r + 8 * half;
    yBase[m * D] = c[r] + bv;
  }
}

// ---------------------------------------------------------------------------
// Kernel 4: per row (b,r):  out = LayerNorm( Y[b,:]*gamma1 + q[b,r,:] )
// one wave per row, 20 floats (5 float4) per lane; wave32 shuffle reduction.
// This kernel carries the ~336 MB q/out stream -> sets the runtime floor.
// ---------------------------------------------------------------------------
__global__ void epilogue(const float* __restrict__ q, const float* __restrict__ Y,
                         const float* __restrict__ g, const float* __restrict__ lw,
                         const float* __restrict__ lb, float* __restrict__ out) {
  const int lane = threadIdx.x & 31;
  const int wave = threadIdx.x >> 5;
  const int row  = blockIdx.x * 8 + wave;           // grid = 8192 blocks -> 65536 rows
  const int b    = row >> 5;                        // batch = row / 32

  const float* qr = q + row * D + lane * 20;
  const float* yr = Y + b * D + lane * 20;
  const float* gr = g + lane * 20;

  float4 z[5];
  float s = 0.f, ss = 0.f;
  #pragma unroll
  for (int i = 0; i < 5; ++i) {
    float4 qv = *reinterpret_cast<const float4*>(qr + i * 4);
    float4 yv = *reinterpret_cast<const float4*>(yr + i * 4);
    float4 gv = *reinterpret_cast<const float4*>(gr + i * 4);
    float4 t;
    t.x = fmaf(yv.x, gv.x, qv.x);
    t.y = fmaf(yv.y, gv.y, qv.y);
    t.z = fmaf(yv.z, gv.z, qv.z);
    t.w = fmaf(yv.w, gv.w, qv.w);
    z[i] = t;
    s  += t.x + t.y + t.z + t.w;
    ss += t.x * t.x + t.y * t.y + t.z * t.z + t.w * t.w;
  }
  #pragma unroll
  for (int off = 16; off; off >>= 1) {
    s  += __shfl_xor(s,  off, 32);
    ss += __shfl_xor(ss, off, 32);
  }
  const float mean = s * (1.0f / D);
  const float var  = ss * (1.0f / D) - mean * mean;
  const float inv  = rsqrtf(var + 1e-5f);

  const float* lwr = lw + lane * 20;
  const float* lbr = lb + lane * 20;
  float* orow = out + row * D + lane * 20;
  #pragma unroll
  for (int i = 0; i < 5; ++i) {
    float4 wv = *reinterpret_cast<const float4*>(lwr + i * 4);
    float4 bv = *reinterpret_cast<const float4*>(lbr + i * 4);
    float4 o;
    o.x = fmaf((z[i].x - mean) * inv, wv.x, bv.x);
    o.y = fmaf((z[i].y - mean) * inv, wv.y, bv.y);
    o.z = fmaf((z[i].z - mean) * inv, wv.z, bv.z);
    o.w = fmaf((z[i].w - mean) * inv, wv.w, bv.w);
    *reinterpret_cast<float4*>(orow + i * 4) = o;
  }
}

// ---------------------------------------------------------------------------
// Launch
// ---------------------------------------------------------------------------
extern "C" void kernel_launch(void* const* d_in, const int* in_sizes, int n_in,
                              void* d_out, int out_size, void* d_ws, size_t ws_size,
                              hipStream_t stream) {
  const float* q      = (const float*)d_in[0];
  const float* k      = (const float*)d_in[1];
  const float* v      = (const float*)d_in[2];
  const float* w_fc   = (const float*)d_in[3];
  const float* b_fc   = (const float*)d_in[4];
  const float* gamma1 = (const float*)d_in[5];
  const float* ln_w   = (const float*)d_in[6];
  const float* ln_b   = (const float*)d_in[7];
  float* out = (float*)d_out;

  char* ws = (char*)d_ws;
  unsigned short* T  = (unsigned short*)ws;                      // 2048*640*2 = 2,621,440 B
  unsigned short* Wb = (unsigned short*)(ws + 2621440);          //  640*640*2 =   819,200 B
  float*          Y  = (float*)(ws + 2621440 + 819200);          // 2048*640*4 = 5,242,880 B

  prep_T  <<<NUM_B / 8,           256, 0, stream>>>(k, v, T);
  conv_W  <<<(D * D) / 256,       256, 0, stream>>>(w_fc, Wb);
  gemm_bf16<<<(NUM_B/16)*(D/16)/8,256, 0, stream>>>(T, Wb, b_fc, Y);
  epilogue<<<NUM_B * LEN_Q / 8,   256, 0, stream>>>(q, Y, gamma1, ln_w, ln_b, out);
}
